// HierarchicalAttentionRecurrent_87840671137941
// MI455X (gfx1250) — matrix operand order repaired and checked by reading, not compiled, not run on hardware
//
#include <hip/hip_runtime.h>
#include <math.h>

// ---------------------------------------------------------------------------
// HierarchicalAttentionRecurrent forward for MI455X (gfx1250, wave32, WMMA).
// All big GEMMs run v_wmma_f32_16x16x32_bf16 on TRANSPOSED tiles so that both
// A and B fragments are contiguous b128 loads from row-major storage:
//   h^T tiles  : A = aw^T [128,256], B = (x*prev)^T (x row-major over d)
//   s^T tiles  : A = ce   [Cp,128],  B = h^T       (h row-major over e)
// att [B,L,C] is never materialized; streaming passes produce (m,den),
// w = sum_c att, prev = sum_c att*pred/C.
// Transcendentals use the gfx1250 hardware v_tanh_f32 / v_exp_f32 paths.
// ---------------------------------------------------------------------------

typedef __attribute__((ext_vector_type(16))) __bf16 v16bf;
typedef __attribute__((ext_vector_type(8)))  __bf16 v8bf;
typedef __attribute__((ext_vector_type(8)))  float  v8f;

#define BATCH 32
#define SEQ   2048
#define DIM   256
#define EMB   100
#define EMBP  128            // E padded to 4 k-steps of 32
#define BL    (BATCH * SEQ)

#define LOG2E 1.44269504088896340736f

#if __has_builtin(__builtin_amdgcn_tanhf)
__device__ __forceinline__ float fast_tanh(float x) { return __builtin_amdgcn_tanhf(x); }
#else
__device__ __forceinline__ float fast_tanh(float x) { return tanhf(x); }
#endif

#if __has_builtin(__builtin_amdgcn_exp2f)
__device__ __forceinline__ float fast_exp(float x) { return __builtin_amdgcn_exp2f(x * LOG2E); }
#else
__device__ __forceinline__ float fast_exp(float x) { return __expf(x); }
#endif

__device__ __forceinline__ v8f wmma_bf16(v16bf a, v16bf b, v8f c) {
  return __builtin_amdgcn_wmma_f32_16x16x32_bf16(
      false, a, false, b, (short)0, c, false, false);
}

__device__ __forceinline__ v8f zero8() {
  v8f z;
#pragma unroll
  for (int i = 0; i < 8; ++i) z[i] = 0.0f;
  return z;
}

__device__ __forceinline__ v16bf cat16(v8bf lo, v8bf hi) {
  return __builtin_shufflevector(lo, hi, 0, 1, 2, 3, 4, 5, 6, 7,
                                         8, 9, 10, 11, 12, 13, 14, 15);
}

// A fragment: lane = row M; elements e -> K = kk*32 + half*16 + e (16 contig). Two b128s.
__device__ __forceinline__ v16bf ld_afrag(const __bf16* __restrict__ rowp,
                                          int kk, int half) {
  v8bf lo = *(const v8bf*)(rowp + kk * 32 + half * 16);
  v8bf hi = *(const v8bf*)(rowp + kk * 32 + half * 16 + 8);
  return cat16(lo, hi);
}

// B fragment: lane = col N; elements e -> K = kk*32 + half*16 + e (16 contig).
__device__ __forceinline__ v16bf ld_bfrag(const __bf16* __restrict__ colp,
                                          int kk, int half) {
  v8bf lo = *(const v8bf*)(colp + kk * 32 + half * 16);
  v8bf hi = *(const v8bf*)(colp + kk * 32 + half * 16 + 8);
  return cat16(lo, hi);
}

// ---------------- small prep kernels ----------------------------------------

__global__ void k_text_mean(const float* __restrict__ x, float* __restrict__ tmean) {
  int b = blockIdx.x, d = threadIdx.x;
  const float* xp = x + (size_t)b * SEQ * DIM + d;
  float acc = 0.0f;
  for (int l = 0; l < SEQ; ++l) acc += xp[(size_t)l * DIM];
  tmean[b * DIM + d] = acc * (1.0f / SEQ);
}

// aw [256,100] -> awT [128,256] bf16 (transposed, zero-padded rows e>=100)
__global__ void k_prep_awT(const float* __restrict__ aw, __bf16* __restrict__ awT) {
  int idx = blockIdx.x * blockDim.x + threadIdx.x;   // 128*256
  int e = idx >> 8, d = idx & 255;
  awT[idx] = (__bf16)((e < EMB) ? aw[d * EMB + e] : 0.0f);
}

// ce [C,100] -> cep [Cp,128] bf16 (row-major, zero-padded)
__global__ void k_prep_cep(const float* __restrict__ ce, __bf16* __restrict__ cep,
                           int C, int Cp) {
  int idx = blockIdx.x * blockDim.x + threadIdx.x;
  if (idx >= Cp * EMBP) return;
  int c = idx >> 7, e = idx & 127;
  cep[idx] = (__bf16)((c < C && e < EMB) ? ce[c * EMB + e] : 0.0f);
}

// ---------------- GEMM 1: h[l,e] = tanh(sum_d x[l,d]*prev[l] * aw[d,e]) ----
// Transposed tiles: M = e (8 tiles), N = l (16 per wave). Per-lane stores are
// one packed b128 per e-tile.
__global__ void k_tanh_gemm(const float* __restrict__ x,
                            const float* __restrict__ prev, int useprev,
                            const __bf16* __restrict__ awT,
                            __bf16* __restrict__ hbuf) {
  int lane = threadIdx.x & 31, wid = threadIdx.x >> 5;
  int half = lane >> 4, mrow = lane & 15;
  int lcol = blockIdx.x * 64 + wid * 16 + mrow;      // global row in [0, BL)
  float pv = useprev ? prev[lcol] : 1.0f;

  v8f acc[8];
#pragma unroll
  for (int n = 0; n < 8; ++n) acc[n] = zero8();

  for (int kk = 0; kk < 8; ++kk) {                   // K = d = 256
    const float* xr = x + (size_t)lcol * DIM + kk * 32 + half * 16;
    v8f f0 = *(const v8f*)xr * pv;
    v8f f1 = *(const v8f*)(xr + 8) * pv;
    v16bf bm = cat16(__builtin_convertvector(f0, v8bf),
                     __builtin_convertvector(f1, v8bf));
#pragma unroll
    for (int et = 0; et < 8; ++et) {
      const __bf16* ar = awT + (size_t)(et * 16 + mrow) * DIM;
      acc[et] = wmma_bf16(ld_afrag(ar, kk, half), bm, acc[et]);
    }
  }

#pragma unroll
  for (int et = 0; et < 8; ++et) {                   // D: row e = et*16+half*8+r
    v8bf o;
#pragma unroll
    for (int r = 0; r < 8; ++r) o[r] = (__bf16)fast_tanh(acc[et][r]);
    *(v8bf*)(hbuf + (size_t)lcol * EMBP + et * 16 + half * 8) = o;
  }
}

// ---------------- scores pass 1: per-(b,c) max & denom over L ---------------
// s^T tiles: A = ce rows (preloaded), B = h rows streamed over 128 l-tiles.
// Per-lane online (m,d)[8]; butterfly-merge across the 16 l-lanes at the end,
// then LDS-merge the 4 waves.
__global__ void k_softmax_stats(const __bf16* __restrict__ hbuf,
                                const __bf16* __restrict__ cep,
                                float* __restrict__ mout,
                                float* __restrict__ dout_, int Cp) {
  int lane = threadIdx.x & 31, wid = threadIdx.x >> 5;
  int half = lane >> 4, mrow = lane & 15;
  int cbase = blockIdx.x * 16, b = blockIdx.y;

  v16bf ak[4];
  const __bf16* ar = cep + (size_t)(cbase + mrow) * EMBP;
#pragma unroll
  for (int kk = 0; kk < 4; ++kk) ak[kk] = ld_afrag(ar, kk, half);

  float m[8], d[8];
#pragma unroll
  for (int r = 0; r < 8; ++r) { m[r] = -__builtin_inff(); d[r] = 0.0f; }

  for (int mt = wid; mt < SEQ / 16; mt += 4) {
    const __bf16* br = hbuf + (size_t)(b * SEQ + mt * 16 + mrow) * EMBP;
    v8f acc = zero8();
#pragma unroll
    for (int kk = 0; kk < 4; ++kk) acc = wmma_bf16(ak[kk], ld_bfrag(br, kk, half), acc);
#pragma unroll
    for (int r = 0; r < 8; ++r) {                    // acc[r] = s[c_r, l=lane's col]
      float v = acc[r];
      float nm = fmaxf(m[r], v);
      d[r] = d[r] * fast_exp(m[r] - nm) + fast_exp(v - nm);
      m[r] = nm;
    }
  }
  // merge the 16 l-columns (lanes with same half bit)
#pragma unroll
  for (int mask = 1; mask < 16; mask <<= 1) {
#pragma unroll
    for (int r = 0; r < 8; ++r) {
      float om = __shfl_xor(m[r], mask, 32);
      float od = __shfl_xor(d[r], mask, 32);
      float nm = fmaxf(m[r], om);
      d[r] = d[r] * fast_exp(m[r] - nm) + od * fast_exp(om - nm);
      m[r] = nm;
    }
  }
  __shared__ float sm[4][16], sd[4][16];
  if (mrow == 0) {
#pragma unroll
    for (int r = 0; r < 8; ++r) { sm[wid][half * 8 + r] = m[r]; sd[wid][half * 8 + r] = d[r]; }
  }
  __syncthreads();
  if (threadIdx.x < 16) {
    float M = -__builtin_inff();
#pragma unroll
    for (int w = 0; w < 4; ++w) M = fmaxf(M, sm[w][threadIdx.x]);
    float Dn = 0.0f;
#pragma unroll
    for (int w = 0; w < 4; ++w) Dn += sd[w][threadIdx.x] * fast_exp(sm[w][threadIdx.x] - M);
    mout[b * Cp + cbase + threadIdx.x]  = M;
    dout_[b * Cp + cbase + threadIdx.x] = Dn;
  }
}

// fuse (m, weight) -> float2 to halve scalar loads in the row-sum pass
__global__ void k_make_mu1(const float* __restrict__ mbuf, const float* __restrict__ den,
                           float2* __restrict__ muf, int C, int Cp) {
  int idx = blockIdx.x * blockDim.x + threadIdx.x;   // BATCH*Cp
  int c = idx % Cp;
  muf[idx] = make_float2(mbuf[idx], (c < C) ? 1.0f / den[idx] : 0.0f);
}

__global__ void k_make_mu2(const float* __restrict__ mbuf, const float* __restrict__ den,
                           const float* __restrict__ pred, float2* __restrict__ muf,
                           int C, int Cp) {
  int idx = blockIdx.x * blockDim.x + threadIdx.x;
  int b = idx / Cp, c = idx % Cp;
  float u = (c < C) ? pred[b * C + c] / (den[idx] * (float)C) : 0.0f;
  muf[idx] = make_float2(mbuf[idx], u);
}

// ---------------- passes 2/3: out[b,l] = sum_c exp(s-m[c]) * u[c] ----------
// B = h rows for this wave's 16 l (preloaded); A = ce streamed over c-tiles.
__global__ void k_weighted_rowsum(const __bf16* __restrict__ hbuf,
                                  const __bf16* __restrict__ cep,
                                  const float2* __restrict__ muf,
                                  float* __restrict__ outw, int Cp) {
  int lane = threadIdx.x & 31, wid = threadIdx.x >> 5;
  int half = lane >> 4, mrow = lane & 15;
  int b = blockIdx.y;
  int ltile = blockIdx.x * 4 + wid;                  // 32 blocks * 4 waves = 128
  const __bf16* br = hbuf + (size_t)(b * SEQ + ltile * 16 + mrow) * EMBP;

  v16bf bk[4];
#pragma unroll
  for (int kk = 0; kk < 4; ++kk) bk[kk] = ld_bfrag(br, kk, half);

  float partial = 0.0f;
  for (int ct = 0; ct < Cp / 16; ++ct) {
    const __bf16* ar = cep + (size_t)(ct * 16 + mrow) * EMBP;
    v8f acc = zero8();
#pragma unroll
    for (int kk = 0; kk < 4; ++kk) acc = wmma_bf16(ld_afrag(ar, kk, half), bk[kk], acc);
    int base = b * Cp + ct * 16 + half * 8;          // c = ct*16 + half*8 + r
#pragma unroll
    for (int r = 0; r < 8; ++r) {
      float2 mu = muf[base + r];
      partial += fast_exp(acc[r] - mu.x) * mu.y;
    }
  }
  partial += __shfl_xor(partial, 16, 32);            // join the two c-halves
  if (lane < 16) outw[b * SEQ + ltile * 16 + lane] = partial;
}

// pooled + feat
__global__ void k_pooled_feat(const float* __restrict__ x, const float* __restrict__ w,
                              const float* __restrict__ prev, int useprev,
                              const float* __restrict__ tmean,
                              float* __restrict__ feat, float invC) {
  int b = blockIdx.x, d = threadIdx.x;
  const float* xp = x + (size_t)b * SEQ * DIM + d;
  float acc = 0.0f;
  for (int l = 0; l < SEQ; ++l) {
    float tw = xp[(size_t)l * DIM];
    if (useprev) tw *= prev[b * SEQ + l];
    acc += w[b * SEQ + l] * tw;
  }
  feat[b * 2 * DIM + d]       = tmean[b * DIM + d];
  feat[b * 2 * DIM + DIM + d] = acc * invC;
}

__global__ void k_dense(const float* __restrict__ feat, const float* __restrict__ dw,
                        const float* __restrict__ db, float* __restrict__ dense) {
  int b = blockIdx.x, n = threadIdx.x;
  const float* f = feat + b * 2 * DIM;
  float acc = db[n];
  for (int k = 0; k < 2 * DIM; ++k) acc += f[k] * dw[k * DIM + n];
  dense[b * DIM + n] = fmaxf(acc, 0.0f);
}

__global__ void k_pred(const float* __restrict__ dense, const float* __restrict__ pw,
                       const float* __restrict__ pb, float* __restrict__ pred, int C) {
  __shared__ float red[256];
  int b = blockIdx.x, tid = threadIdx.x;
  const float* dn = dense + b * DIM;
  float lg[4];
  int cnt = 0;
  for (int c = tid; c < C; c += 256, ++cnt) {
    float acc = pb[c];
    for (int k = 0; k < DIM; ++k) acc += dn[k] * pw[k * C + c];
    lg[cnt] = acc;
  }
  float lm = -__builtin_inff();
  for (int i = 0; i < cnt; ++i) lm = fmaxf(lm, lg[i]);
  red[tid] = lm; __syncthreads();
  for (int s = 128; s > 0; s >>= 1) { if (tid < s) red[tid] = fmaxf(red[tid], red[tid + s]); __syncthreads(); }
  float gmax = red[0]; __syncthreads();
  float ls = 0.0f;
  for (int i = 0; i < cnt; ++i) ls += fast_exp(lg[i] - gmax);
  red[tid] = ls; __syncthreads();
  for (int s = 128; s > 0; s >>= 1) { if (tid < s) red[tid] += red[tid + s]; __syncthreads(); }
  float gsum = red[0];
  cnt = 0;
  for (int c = tid; c < C; c += 256, ++cnt)
    pred[b * C + c] = fast_exp(lg[cnt] - gmax) / gsum;
}

// ---------------------------------------------------------------------------

extern "C" void kernel_launch(void* const* d_in, const int* in_sizes, int n_in,
                              void* d_out, int out_size, void* d_ws, size_t ws_size,
                              hipStream_t stream) {
  (void)in_sizes; (void)n_in; (void)out_size; (void)ws_size;
  const float* x = (const float*)d_in[0];

  char* p = (char*)d_ws;
  auto carve = [&](size_t bytes) { char* r = p; p += (bytes + 255) & ~(size_t)255; return (void*)r; };
  float*  tmean = (float*) carve((size_t)BATCH * DIM * 4);
  __bf16* hbuf  = (__bf16*)carve((size_t)BL * EMBP * 2);       // 16.8 MB
  __bf16* awT   = (__bf16*)carve((size_t)EMBP * DIM * 2);
  __bf16* cep   = (__bf16*)carve((size_t)1024 * EMBP * 2);
  float*  mbuf  = (float*) carve((size_t)BATCH * 1024 * 4);
  float*  dbuf  = (float*) carve((size_t)BATCH * 1024 * 4);
  float2* muf   = (float2*)carve((size_t)BATCH * 1024 * 8);
  float*  wbuf  = (float*) carve((size_t)BATCH * SEQ * 4);
  float*  prevb = (float*) carve((size_t)BATCH * SEQ * 4);
  float*  feat  = (float*) carve((size_t)BATCH * 2 * DIM * 4);
  float*  dense = (float*) carve((size_t)BATCH * DIM * 4);

  k_text_mean<<<BATCH, DIM, 0, stream>>>(x, tmean);

  const int Cs[3]  = {21, 191, 1009};
  const int Cps[3] = {32, 192, 1024};
  float* out = (float*)d_out;
  size_t off = 0;

  for (int h = 0; h < 3; ++h) {
    const float* ce = (const float*)d_in[1 + 6 * h + 0];
    const float* aw = (const float*)d_in[1 + 6 * h + 1];
    const float* dw = (const float*)d_in[1 + 6 * h + 2];
    const float* db = (const float*)d_in[1 + 6 * h + 3];
    const float* pw = (const float*)d_in[1 + 6 * h + 4];
    const float* pb = (const float*)d_in[1 + 6 * h + 5];
    int C = Cs[h], Cp = Cps[h];
    int useprev = (h > 0) ? 1 : 0;

    k_prep_awT<<<(EMBP * DIM) / 256, 256, 0, stream>>>(aw, awT);
    k_prep_cep<<<(Cp * EMBP + 255) / 256, 256, 0, stream>>>(ce, cep, C, Cp);

    k_tanh_gemm<<<BL / 64, 128, 0, stream>>>(x, prevb, useprev, awT, hbuf);

    k_softmax_stats<<<dim3(Cp / 16, BATCH), 128, 0, stream>>>(hbuf, cep, mbuf, dbuf, Cp);

    k_make_mu1<<<(BATCH * Cp) / 256, 256, 0, stream>>>(mbuf, dbuf, muf, C, Cp);
    k_weighted_rowsum<<<dim3(SEQ / 64, BATCH), 128, 0, stream>>>(hbuf, cep, muf, wbuf, Cp);

    k_pooled_feat<<<BATCH, DIM, 0, stream>>>(x, wbuf, prevb, useprev, tmean, feat, 1.0f / (float)C);
    k_dense<<<BATCH, DIM, 0, stream>>>(feat, dw, db, dense);
    k_pred<<<BATCH, 256, 0, stream>>>(dense, pw, pb, out + off, C);

    if (h < 2) {
      k_make_mu2<<<(BATCH * Cp) / 256, 256, 0, stream>>>(mbuf, dbuf, out + off, muf, C, Cp);
      k_weighted_rowsum<<<dim3(SEQ / 64, BATCH), 128, 0, stream>>>(hbuf, cep, muf, prevb, Cp);
    }
    off += (size_t)BATCH * C;
  }
}
